// CompilablemLSTMBlock_70428873719997
// MI455X (gfx1250) — compile-verified, matrix operand order checked by reading, not run on hardware
//
#include <hip/hip_runtime.h>
#include <hip/hip_bf16.h>
#include <math.h>

#define B_   4
#define S_   2048
#define D_   1024
#define H_   16
#define DH_  64
#define NROW (B_ * S_)        // 8192 rows
#define NQKV (3 * H_ * DH_)   // 3072 cols

typedef __attribute__((ext_vector_type(8)))  __bf16 v8bf;
typedef __attribute__((ext_vector_type(16))) __bf16 v16bf;
typedef __attribute__((ext_vector_type(8)))  float  v8f;

__device__ __forceinline__ v8f vzero() {
  v8f z;
  #pragma unroll
  for (int i = 0; i < 8; ++i) z[i] = 0.f;
  return z;
}

// Load one 16x32 bf16 WMMA operand fragment from a row-major matrix
// (rows along M/N, contiguous along K). Per CDNA5 ISA 16-bit A layout:
// lane L holds row (L&15); VGPR0-3 hold K = kb..kb+7, VGPR4-7 hold
// K = kb+16..kb+23, with kb = (L>>4)*8. Two 16-byte loads per lane.
__device__ __forceinline__ v16bf load_frag(const __bf16* base, int ld,
                                           int row0, int k0, int lane) {
  int r = row0 + (lane & 15);
  int k = k0 + ((lane >> 4) << 3);
  const __bf16* p = base + (size_t)r * ld + k;
  v8bf lo = *reinterpret_cast<const v8bf*>(p);
  v8bf hi = *reinterpret_cast<const v8bf*>(p + 16);
  return __builtin_shufflevector(lo, hi, 0,1,2,3,4,5,6,7,8,9,10,11,12,13,14,15);
}

// 16-byte global -> LDS async copy (GLOBAL_LOAD_ASYNC_TO_LDS_B128, ASYNCcnt).
// LDS operand is the per-lane LDS byte offset (flat ptr low 32 bits).
__device__ __forceinline__ void async_copy_b128(const __bf16* gsrc, __bf16* ldst) {
  unsigned int loff = (unsigned int)(uintptr_t)ldst;
  asm volatile("global_load_async_to_lds_b128 %0, %1, off"
               :: "v"(loff), "v"(gsrc) : "memory");
}
__device__ __forceinline__ void wait_asynccnt0() {
  asm volatile("s_wait_asynccnt 0x0" ::: "memory");
}

// ---------------- RMSNorm: x (f32) -> xn (bf16) ----------------
__global__ __launch_bounds__(256) void k_rmsnorm(const float* __restrict__ x,
                                                 const float* __restrict__ w,
                                                 __bf16* __restrict__ xn) {
  int row = blockIdx.x;
  const float* xr = x + (size_t)row * D_;
  float v[4]; float s = 0.f;
  #pragma unroll
  for (int i = 0; i < 4; ++i) { v[i] = xr[threadIdx.x + 256 * i]; s += v[i] * v[i]; }
  __shared__ float red[256];
  red[threadIdx.x] = s; __syncthreads();
  for (int off = 128; off > 0; off >>= 1) {
    if (threadIdx.x < off) red[threadIdx.x] += red[threadIdx.x + off];
    __syncthreads();
  }
  float scale = rsqrtf(red[0] * (1.f / D_) + 1e-6f);
  #pragma unroll
  for (int i = 0; i < 4; ++i) {
    int d = threadIdx.x + 256 * i;
    xn[(size_t)row * D_ + d] = (__bf16)(w[d] * v[i] * scale);
  }
}

// ------------- weight transpose + f32->bf16: wT[n*K+k] = w[k*N+n] -------------
__global__ __launch_bounds__(256) void k_transpose_cvt(const float* __restrict__ w,
                                                       __bf16* __restrict__ wT,
                                                       int K, int N) {
  size_t idx = (size_t)blockIdx.x * 256 + threadIdx.x;
  if (idx >= (size_t)K * N) return;
  int n = (int)(idx / K), k = (int)(idx % K);
  wT[idx] = (__bf16)w[(size_t)k * N + n];
}

// ------------- gates: sigmoid(15*tanh((xn @ w_gate)/15)) -------------
__global__ __launch_bounds__(64) void k_gates(const __bf16* __restrict__ xn,
                                              const float* __restrict__ wg,
                                              float* __restrict__ gates) {
  int row = blockIdx.x, g = threadIdx.x;
  if (g >= 48) return;
  const __bf16* xr = xn + (size_t)row * D_;
  float acc = 0.f;
  for (int d = 0; d < D_; ++d) acc += (float)xr[d] * wg[(size_t)d * 48 + g];
  float t = 15.f * tanhf(acc * (1.f / 15.f));
  gates[(size_t)row * 48 + g] = 1.f / (1.f + expf(-t));
}

// ------------- bf16 WMMA GEMM: C[M,N] = A[M,K] @ BT[N,K]^T -------------
// block tile 128x128, 8 waves (2x4), wave tile 64x32 (4x2 wmma tiles)
__global__ __launch_bounds__(256) void k_gemm_bf16(const __bf16* __restrict__ A,
                                                   const __bf16* __restrict__ BT,
                                                   int K, int N,
                                                   __bf16* __restrict__ outB,
                                                   float* __restrict__ outF,
                                                   const float* __restrict__ residual) {
  int lane = threadIdx.x & 31, wv = threadIdx.x >> 5;
  int wm = wv >> 2, wn = wv & 3;
  int m0 = blockIdx.y * 128 + wm * 64;
  int n0 = blockIdx.x * 128 + wn * 32;

  v8f acc[4][2];
  #pragma unroll
  for (int mt = 0; mt < 4; ++mt)
    #pragma unroll
    for (int nt = 0; nt < 2; ++nt) acc[mt][nt] = vzero();

  for (int k0 = 0; k0 < K; k0 += 32) {
    v16bf a[4], bfr[2];
    #pragma unroll
    for (int mt = 0; mt < 4; ++mt) a[mt] = load_frag(A, K, m0 + mt * 16, k0, lane);
    #pragma unroll
    for (int nt = 0; nt < 2; ++nt) bfr[nt] = load_frag(BT, K, n0 + nt * 16, k0, lane);
    // prefetch the next-but-one K slice (global_prefetch_b8; speculative OOB is dropped)
    {
      int kp = k0 + 64;
      int r  = (lane & 15);
      #pragma unroll
      for (int mt = 0; mt < 4; ++mt)
        __builtin_prefetch(A + (size_t)(m0 + mt * 16 + r) * K + kp, 0, 3);
      #pragma unroll
      for (int nt = 0; nt < 2; ++nt)
        __builtin_prefetch(BT + (size_t)(n0 + nt * 16 + r) * K + kp, 0, 3);
    }
    #pragma unroll
    for (int mt = 0; mt < 4; ++mt)
      #pragma unroll
      for (int nt = 0; nt < 2; ++nt)
        acc[mt][nt] = __builtin_amdgcn_wmma_f32_16x16x32_bf16(
            false, a[mt], false, bfr[nt], (short)0, acc[mt][nt], false, false);
  }

  // C-layout: m = mt*16 + r + 8*(lane>>4), n = nt*16 + (lane&15)
  const int roff = (lane >> 4) << 3;
  const int cn   = lane & 15;
  if (outF) {
    #pragma unroll
    for (int mt = 0; mt < 4; ++mt)
      #pragma unroll
      for (int nt = 0; nt < 2; ++nt)
        #pragma unroll
        for (int r = 0; r < 8; ++r) {
          size_t o = (size_t)(m0 + mt * 16 + roff + r) * N + (n0 + nt * 16 + cn);
          outF[o] = residual[o] + acc[mt][nt][r];
        }
  } else {
    #pragma unroll
    for (int mt = 0; mt < 4; ++mt)
      #pragma unroll
      for (int nt = 0; nt < 2; ++nt)
        #pragma unroll
        for (int r = 0; r < 8; ++r) {
          size_t o = (size_t)(m0 + mt * 16 + roff + r) * N + (n0 + nt * 16 + cn);
          outB[o] = (__bf16)acc[mt][nt][r];
        }
  }
}

// ------------- chunkwise-parallel mLSTM scan, one block per (b,h) -------------
// chunk L=64: S=Q K^T ; P = S .* exp(la_t - la_u) * i_u (u<=t)
// H = P@V + diag(e^{la_t}) Q @ state ; state = e^{la_L} state + (K diag(w))^T V
__global__ __launch_bounds__(256) void k_scan(const __bf16* __restrict__ qkv,
                                              const float* __restrict__ gates,
                                              const float* __restrict__ hs0,
                                              __bf16* __restrict__ hout) {
  const int bh = blockIdx.x, b = bh >> 4, h = bh & 15;
  const int tid = threadIdx.x, lane = tid & 31, wv = tid >> 5;
  const int mt = wv >> 1, nt0 = (wv & 1) * 2;   // 8 waves cover 4x4 16x16 tiles

  __shared__ __align__(16) __bf16 Qc[64][72];   // [t][d]
  __shared__ __align__(16) __bf16 Kc[64][72];   // [u][d]
  __shared__ __align__(16) __bf16 Vt[64][72];   // [e][u]  (V transposed)
  __shared__ __align__(16) __bf16 KtW[64][72];  // [d][u] = k[u][d]*w_u
  __shared__ __align__(16) __bf16 Pm[64][72];   // [t][u]
  __shared__ __align__(16) __bf16 STbf[64][72]; // [e][d]  (state^T, bf16 mirror)
  __shared__ float la_s[64], ig_s[64], og_s[64], lf_s[64];

  // persistent f32 state in WMMA C-layout registers: tile (e-tile=mt, d-tile=nt0+j)
  v8f st[2];
  const float* hsb = hs0 + (size_t)bh * DH_ * DH_;
  #pragma unroll
  for (int j = 0; j < 2; ++j)
    #pragma unroll
    for (int r = 0; r < 8; ++r) {
      int e = mt * 16 + ((lane >> 4) << 3) + r;
      int d = (nt0 + j) * 16 + (lane & 15);
      float sv = hsb[d * 64 + e];   // state[d][e]
      st[j][r] = sv;
      STbf[e][d] = (__bf16)sv;
    }
  __syncthreads();

  for (int c = 0; c < S_ / 64; ++c) {
    const size_t rowBase = (size_t)(b * S_ + c * 64);

    // --- Q/K chunk staging via async global->LDS (ASYNCcnt), 16B per lane ---
    // 64 rows x 128B per tile = 512 b128 transfers per tile; 256 threads x 2.
    #pragma unroll
    for (int it = 0; it < 2; ++it) {
      int i   = tid + 256 * it;
      int row = i >> 3, seg = (i & 7) * 8;
      const __bf16* g = qkv + (rowBase + row) * NQKV + h * DH_ + seg;
      async_copy_b128(g,            &Qc[row][seg]);
      async_copy_b128(g + H_ * DH_, &Kc[row][seg]);
    }
    // V needs a transpose into LDS: VALU path.
    for (int idx = tid; idx < 64 * 64; idx += 256) {
      int tl = idx >> 6, d = idx & 63;
      Vt[d][tl] = qkv[(rowBase + tl) * NQKV + 2 * H_ * DH_ + h * DH_ + d];
    }
    if (tid < 64) {
      const float* gp = gates + (rowBase + tid) * 48 + h;
      ig_s[tid] = gp[0];
      lf_s[tid] = logf(fmaxf(gp[H_], 1e-30f));
      og_s[tid] = gp[2 * H_];
    }
    wait_asynccnt0();
    __syncthreads();
    if (tid == 0) {   // inclusive log-decay cumsum
      float run = 0.f;
      for (int t = 0; t < 64; ++t) { run += lf_s[t]; la_s[t] = run; }
    }
    __syncthreads();
    const float laL = la_s[63];

    // KtW[d][u] = k[u][d] * exp(laL - la_u) * i_u
    for (int idx = tid; idx < 64 * 64; idx += 256) {
      int d = idx >> 6, u = idx & 63;
      float w = expf(laL - la_s[u]) * ig_s[u];
      KtW[d][u] = (__bf16)((float)Kc[u][d] * w);
    }

    // G1: S = Q @ K^T  (A=Qc rows t, B cols u from Kc rows u)
    v8f sacc[2]; sacc[0] = vzero(); sacc[1] = vzero();
    for (int ks = 0; ks < 64; ks += 32) {
      v16bf af = load_frag(&Qc[0][0], 72, mt * 16, ks, lane);
      #pragma unroll
      for (int j = 0; j < 2; ++j) {
        v16bf bf = load_frag(&Kc[0][0], 72, (nt0 + j) * 16, ks, lane);
        sacc[j] = __builtin_amdgcn_wmma_f32_16x16x32_bf16(
            false, af, false, bf, (short)0, sacc[j], false, false);
      }
    }
    #pragma unroll
    for (int j = 0; j < 2; ++j)
      #pragma unroll
      for (int r = 0; r < 8; ++r) {
        int t = mt * 16 + ((lane >> 4) << 3) + r;
        int u = (nt0 + j) * 16 + (lane & 15);
        float p = (u <= t) ? sacc[j][r] * expf(la_s[t] - la_s[u]) * ig_s[u] : 0.f;
        Pm[t][u] = (__bf16)p;
      }
    __syncthreads();

    // scale Qc in place by exp(la_t) for the inter-chunk term
    for (int idx = tid; idx < 64 * 64; idx += 256) {
      int t = idx >> 6, d = idx & 63;
      Qc[t][d] = (__bf16)((float)Qc[t][d] * expf(la_s[t]));
    }
    __syncthreads();

    // G2+G3: H = Pm @ V + Qs @ state   (both output (t,e))
    v8f hacc[2]; hacc[0] = vzero(); hacc[1] = vzero();
    for (int ks = 0; ks < 64; ks += 32) {
      v16bf af = load_frag(&Pm[0][0], 72, mt * 16, ks, lane);
      #pragma unroll
      for (int j = 0; j < 2; ++j) {
        v16bf bf = load_frag(&Vt[0][0], 72, (nt0 + j) * 16, ks, lane);
        hacc[j] = __builtin_amdgcn_wmma_f32_16x16x32_bf16(
            false, af, false, bf, (short)0, hacc[j], false, false);
      }
    }
    for (int ks = 0; ks < 64; ks += 32) {
      v16bf af = load_frag(&Qc[0][0], 72, mt * 16, ks, lane);
      #pragma unroll
      for (int j = 0; j < 2; ++j) {
        v16bf bf = load_frag(&STbf[0][0], 72, (nt0 + j) * 16, ks, lane);
        hacc[j] = __builtin_amdgcn_wmma_f32_16x16x32_bf16(
            false, af, false, bf, (short)0, hacc[j], false, false);
      }
    }
    #pragma unroll
    for (int j = 0; j < 2; ++j)
      #pragma unroll
      for (int r = 0; r < 8; ++r) {
        int t = mt * 16 + ((lane >> 4) << 3) + r;
        int e = (nt0 + j) * 16 + (lane & 15);
        float v = hacc[j][r] * og_s[t];
        hout[(rowBase + t) * (size_t)D_ + h * DH_ + e] = (__bf16)v;
      }
    __syncthreads();   // all STbf reads done before update writes

    // G4: state update, output (e,d): A=Vt rows e, B cols d from KtW rows d
    v8f uacc[2]; uacc[0] = vzero(); uacc[1] = vzero();
    for (int ks = 0; ks < 64; ks += 32) {
      v16bf af = load_frag(&Vt[0][0], 72, mt * 16, ks, lane);
      #pragma unroll
      for (int j = 0; j < 2; ++j) {
        v16bf bf = load_frag(&KtW[0][0], 72, (nt0 + j) * 16, ks, lane);
        uacc[j] = __builtin_amdgcn_wmma_f32_16x16x32_bf16(
            false, af, false, bf, (short)0, uacc[j], false, false);
      }
    }
    const float aL = expf(laL);
    #pragma unroll
    for (int j = 0; j < 2; ++j)
      #pragma unroll
      for (int r = 0; r < 8; ++r) {
        int e = mt * 16 + ((lane >> 4) << 3) + r;
        int d = (nt0 + j) * 16 + (lane & 15);
        st[j][r] = aL * st[j][r] + uacc[j][r];
        STbf[e][d] = (__bf16)st[j][r];
      }
    __syncthreads();   // protect LDS tiles before next chunk's loads
  }
}

extern "C" void kernel_launch(void* const* d_in, const int* in_sizes, int n_in,
                              void* d_out, int out_size, void* d_ws, size_t ws_size,
                              hipStream_t stream) {
  (void)in_sizes; (void)n_in; (void)out_size; (void)ws_size;
  const float* x     = (const float*)d_in[0];
  const float* hs    = (const float*)d_in[1];
  const float* wrms  = (const float*)d_in[2];
  const float* wqkv  = (const float*)d_in[3];
  const float* wgate = (const float*)d_in[4];
  const float* wout  = (const float*)d_in[5];
  float* out = (float*)d_out;

  char* wsp = (char*)d_ws;
  auto alloc = [&](size_t bytes) {
    char* p = wsp; wsp += (bytes + 255) & ~(size_t)255; return p;
  };
  __bf16* xn    = (__bf16*)alloc((size_t)NROW * D_ * 2);     // 16 MB
  __bf16* wqkvT = (__bf16*)alloc((size_t)NQKV * D_ * 2);     //  6 MB
  __bf16* woutT = (__bf16*)alloc((size_t)D_ * D_ * 2);       //  2 MB
  __bf16* qkvB  = (__bf16*)alloc((size_t)NROW * NQKV * 2);   // 48 MB
  float*  gts   = (float*) alloc((size_t)NROW * 48 * 4);     // 1.5 MB
  __bf16* hbuf  = (__bf16*)alloc((size_t)NROW * D_ * 2);     // 16 MB

  k_transpose_cvt<<<(NQKV * D_ + 255) / 256, 256, 0, stream>>>(wqkv, wqkvT, D_, NQKV);
  k_transpose_cvt<<<(D_ * D_ + 255) / 256, 256, 0, stream>>>(wout, woutT, D_, D_);
  k_rmsnorm<<<NROW, 256, 0, stream>>>(x, wrms, xn);
  k_gates<<<NROW, 64, 0, stream>>>(xn, wgate, gts);
  k_gemm_bf16<<<dim3(NQKV / 128, NROW / 128), 256, 0, stream>>>(
      xn, wqkvT, D_, NQKV, qkvB, nullptr, nullptr);
  k_scan<<<B_ * H_, 256, 0, stream>>>(qkvB, gts, hs, hbuf);
  k_gemm_bf16<<<dim3(D_ / 128, NROW / 128), 256, 0, stream>>>(
      hbuf, woutT, D_, D_, nullptr, out, x);
}
